// Attention_88390426952056
// MI455X (gfx1250) — matrix operand order repaired and checked by reading, not compiled, not run on hardware
//
#include <hip/hip_runtime.h>
#include <hip/hip_bf16.h>
#include <stdint.h>

// ---------------------------------------------------------------------------
// Bahdanau attention on MI455X (gfx1250, wave32).
// Dominant GEMM [65536x512]x[512x512] with v_wmma_f32_16x16x32_bf16
// (bf16 in / f32 acc), fused tanh + V-dot epilogue (no [B,T,U] tensor).
// B-tiles streamed into LDS with GLOBAL_LOAD_ASYNC_TO_LDS_B128 (ASYNCcnt)
// and double-buffered against the WMMA loop.
// ---------------------------------------------------------------------------

typedef __attribute__((ext_vector_type(16))) __bf16        v16bf;
typedef __attribute__((ext_vector_type(8)))  float         v8f;
typedef __attribute__((ext_vector_type(4)))  float         f4;
typedef __attribute__((ext_vector_type(4)))  unsigned int  u4;
typedef __attribute__((ext_vector_type(4)))  unsigned short us4;

#define NB 32
#define NT 2048
#define ND 512
#define NU 512

__device__ __forceinline__ unsigned short f32_to_bf16_rne(float f) {
    union { float f; uint32_t u; } c; c.f = f;
    uint32_t u = c.u;
    uint32_t r = u + 0x7FFFu + ((u >> 16) & 1u);   // round-to-nearest-even
    return (unsigned short)(r >> 16);
}

template <int OFF>
__device__ __forceinline__ float swz_xor(float x) {
    union { float f; int i; } c; c.f = x;
    c.i = __builtin_amdgcn_ds_swizzle(c.i, OFF);   // group-of-32 xor swizzle
    return c.f;
}

// async global -> LDS, 16 bytes per lane, tracked by ASYNCcnt
__device__ __forceinline__ void async_ld_b128(uint32_t lds_off, const void* g) {
    asm volatile("global_load_async_to_lds_b128 %0, %1, off"
                 :: "v"(lds_off), "v"(g)
                 : "memory");
}
__device__ __forceinline__ void wait_async0() {
    asm volatile("s_wait_asynccnt 0x0" ::: "memory");
}

// ---------------------------------------------------------------------------
// K0: W1 [k][n] fp32  ->  Wt [n][k] bf16 (transposed, 512KB, L2-resident)
// ---------------------------------------------------------------------------
__global__ void k_w1_to_bf16_t(const float* __restrict__ W1,
                               unsigned short* __restrict__ Wt) {
    int o = blockIdx.x * 256 + threadIdx.x;        // 0 .. 262143
    int n = o >> 9, k = o & 511;
    Wt[o] = f32_to_bf16_rne(W1[k * NU + n]);       // coalesced writes
}

// ---------------------------------------------------------------------------
// K1: ph[b][u] = hidden[b] . W2[:,u] + b1[u] + b2[u]   (folds both biases)
// ---------------------------------------------------------------------------
__global__ void k_hidden_proj(const float* __restrict__ hidden,
                              const float* __restrict__ W2,
                              const float* __restrict__ b1,
                              const float* __restrict__ b2,
                              float* __restrict__ ph) {
    __shared__ float h[ND];
    const int b = blockIdx.x, tid = threadIdx.x;
    h[tid]       = hidden[b * ND + tid];
    h[tid + 256] = hidden[b * ND + tid + 256];
    __syncthreads();
    const int u0 = tid * 2;
    float a0 = 0.f, a1 = 0.f;
    for (int d = 0; d < ND; ++d) {
        const float hv = h[d];
        const float2 w = *(const float2*)(W2 + d * NU + u0);
        a0 += hv * w.x; a1 += hv * w.y;
    }
    ph[b * NU + u0]     = a0 + b1[u0]     + b2[u0];
    ph[b * NU + u0 + 1] = a1 + b1[u0 + 1] + b2[u0 + 1];
}

// ---------------------------------------------------------------------------
// K2: fused  logits[b,t] = V . tanh(features[b,t,:] @ W1 + ph[b]) + bv
// 256 threads = 8 waves; WG tile = 64 rows x 512 cols; wave = 4 Mx4 N tiles.
// Double-buffered LDS staging: B via async DMA, A via convert+ds_store.
// ---------------------------------------------------------------------------
__global__ __launch_bounds__(256)
void k_scores(const float* __restrict__ features,
              const unsigned short* __restrict__ Wt,   // [n][k] bf16
              const float* __restrict__ ph,
              const float* __restrict__ V,
              const float* __restrict__ bv,
              float* __restrict__ logits) {
    __shared__ unsigned short Abf[2][64 * 32];   // [row][k] bf16, 2x4KB
    __shared__ unsigned short Bbf[2][512 * 32];  // [n][k]  bf16, 2x32KB
    __shared__ float part[64 * 8];               // per-wave row partials, 2KB

    const int tid  = threadIdx.x;
    const int w    = tid >> 5;                // wave 0..7 -> N block of 64
    const int lane = tid & 31;
    const int m16  = lane & 15;
    const int kh   = lane >> 4;               // lane half (WMMA layouts)
    const int b    = blockIdx.y;
    const int t0   = blockIdx.x * 64;

    v8f acc[4][4];
    #pragma unroll
    for (int i = 0; i < 4; ++i)
        #pragma unroll
        for (int j = 0; j < 4; ++j)
            acc[i][j] = (v8f){0.f,0.f,0.f,0.f,0.f,0.f,0.f,0.f};

    const size_t featBase = ((size_t)b * NT + t0) * ND;

    // ---- staging helpers ---------------------------------------------------
    auto stageB = [&](int k0, int buf) {     // 512 n x 32 K bf16, async DMA
        #pragma unroll
        for (int it = 0; it < 8; ++it) {
            int idx = tid + it * 256;        // 0..2047
            int n = idx >> 2, p = idx & 3;   // 4 x 16B per n-row
            async_ld_b128((uint32_t)(size_t)&Bbf[buf][n * 32 + p * 8],
                          Wt + (size_t)n * ND + k0 + p * 8);
        }
    };
    auto stageA = [&](int k0, int buf) {     // 64 rows x 32 K fp32 -> bf16
        #pragma unroll
        for (int it = 0; it < 2; ++it) {
            int idx = tid + it * 256;        // 0..511
            int row = idx >> 3, j = idx & 7; // 8 float4 per row
            f4 f = *(const f4*)(features + featBase + (size_t)row * ND + k0 + j * 4);
            us4 p = (us4){ f32_to_bf16_rne(f.x), f32_to_bf16_rne(f.y),
                           f32_to_bf16_rne(f.z), f32_to_bf16_rne(f.w) };
            *(us4*)(&Abf[buf][row * 32 + j * 4]) = p;
        }
    };

    // ---- prologue: fill buffer 0 ------------------------------------------
    stageB(0, 0);
    stageA(0, 0);
    wait_async0();
    __syncthreads();

    for (int c = 0; c < 16; ++c) {
        const int cur = c & 1, nxt = cur ^ 1;
        // kick off next chunk's fills while this chunk computes
        if (c + 1 < 16) {
            stageB((c + 1) * 32, nxt);
            stageA((c + 1) * 32, nxt);
        }

        // ---- gather fragments (ISA 7.12.2 16-bit layouts) ------------------
        union AB { u4 q[2]; v16bf v; };
        AB afr[4], bfr[4];
        #pragma unroll
        for (int mt = 0; mt < 4; ++mt) {
            const unsigned short* base = &Abf[cur][(mt * 16 + m16) * 32];
            afr[mt].q[0] = *(const u4*)(base + kh * 8);        // K=kh*8..+7
            afr[mt].q[1] = *(const u4*)(base + 16 + kh * 8);   // K=16+kh*8..+7
        }
        #pragma unroll
        for (int nt = 0; nt < 4; ++nt) {
            const unsigned short* base = &Bbf[cur][(w * 64 + nt * 16 + m16) * 32];
            bfr[nt].q[0] = *(const u4*)(base + kh * 8);        // K=kh*8..+7
            bfr[nt].q[1] = *(const u4*)(base + 16 + kh * 8);   // K=16+kh*8..+7
        }

        // ---- 16 WMMAs covering the whole K=32 chunk ------------------------
        #pragma unroll
        for (int mt = 0; mt < 4; ++mt)
            #pragma unroll
            for (int nt = 0; nt < 4; ++nt)
                acc[mt][nt] = __builtin_amdgcn_wmma_f32_16x16x32_bf16(
                    false, afr[mt].v, false, bfr[nt].v,
                    (short)0, acc[mt][nt], false, false);

        // my async fills must land before anyone reads buffer `nxt`
        wait_async0();
        __syncthreads();
    }

    // ---- epilogue: tanh(c + ph) * V, reduce over U deterministically -------
    float vv[4], pv[4];
    #pragma unroll
    for (int nt = 0; nt < 4; ++nt) {
        const int col = w * 64 + nt * 16 + m16;
        vv[nt] = V[col];
        pv[nt] = ph[b * NU + col];
    }
    #pragma unroll
    for (int mt = 0; mt < 4; ++mt) {
        #pragma unroll
        for (int r = 0; r < 8; ++r) {
            float p = 0.f;
            #pragma unroll
            for (int nt = 0; nt < 4; ++nt) {
                const float s = tanhf(acc[mt][nt][r] + pv[nt]);
                p += s * vv[nt];
            }
            // sum over the 16 lanes of this half-wave (xor swizzles 1,2,4,8)
            p += swz_xor<0x041F>(p);
            p += swz_xor<0x081F>(p);
            p += swz_xor<0x101F>(p);
            p += swz_xor<0x201F>(p);
            const int row = mt * 16 + r + kh * 8;   // C layout: hi lanes = M+8
            if (m16 == 0) part[row * 8 + w] = p;
        }
    }
    __syncthreads();
    if (tid < 64) {
        float l = *bv;
        #pragma unroll
        for (int q = 0; q < 8; ++q) l += part[tid * 8 + q];
        logits[b * NT + t0 + tid] = l;
    }
}

// ---------------------------------------------------------------------------
// K3: per-batch softmax over T + context = sum_t attn * features  (fused)
// ---------------------------------------------------------------------------
__global__ void k_softmax_ctx(const float* __restrict__ features,
                              const float* __restrict__ logits,
                              float* __restrict__ out) {
    __shared__ float attn_s[NT];
    __shared__ float red[256];
    const int b = blockIdx.x, tid = threadIdx.x;

    float l[8];
    float lmax = -3.402823466e38f;
    #pragma unroll
    for (int j = 0; j < 8; ++j) {
        l[j] = logits[b * NT + tid + j * 256];
        lmax = fmaxf(lmax, l[j]);
    }
    red[tid] = lmax; __syncthreads();
    for (int s = 128; s > 0; s >>= 1) {
        if (tid < s) red[tid] = fmaxf(red[tid], red[tid + s]);
        __syncthreads();
    }
    const float m = red[0]; __syncthreads();

    float lsum = 0.f;
    #pragma unroll
    for (int j = 0; j < 8; ++j) {
        const float e = __expf(l[j] - m);
        attn_s[tid + j * 256] = e;
        lsum += e;
    }
    red[tid] = lsum; __syncthreads();
    for (int s = 128; s > 0; s >>= 1) {
        if (tid < s) red[tid] += red[tid + s];
        __syncthreads();
    }
    const float inv = 1.0f / red[0];

    float* out_ctx  = out;                 // [32][512]
    float* out_attn = out + NB * ND;       // [32][2048]
    #pragma unroll
    for (int j = 0; j < 8; ++j) {
        const int t = tid + j * 256;
        const float a = attn_s[t] * inv;
        attn_s[t] = a;
        out_attn[b * NT + t] = a;
    }
    __syncthreads();

    // context: each thread owns 2 consecutive D columns, streams all T rows
    const float2* f2 = (const float2*)features;
    const size_t base = (size_t)b * NT * (ND / 2);
    float ax = 0.f, ay = 0.f;
    for (int t = 0; t < NT; ++t) {
        const float a = attn_s[t];
        const float2 f = f2[base + (size_t)t * (ND / 2) + tid];
        ax += a * f.x; ay += a * f.y;
    }
    out_ctx[b * ND + tid * 2]     = ax;
    out_ctx[b * ND + tid * 2 + 1] = ay;
}

// ---------------------------------------------------------------------------
extern "C" void kernel_launch(void* const* d_in, const int* in_sizes, int n_in,
                              void* d_out, int out_size, void* d_ws, size_t ws_size,
                              hipStream_t stream) {
    const float* features = (const float*)d_in[0];
    const float* hidden   = (const float*)d_in[1];
    const float* W1       = (const float*)d_in[2];
    const float* b1       = (const float*)d_in[3];
    const float* W2       = (const float*)d_in[4];
    const float* b2       = (const float*)d_in[5];
    const float* V        = (const float*)d_in[6];
    const float* bv       = (const float*)d_in[7];

    // workspace layout: logits (256KB) | ph (64KB) | Wt bf16 (512KB)
    float* logits = (float*)d_ws;
    float* ph     = logits + NB * NT;
    unsigned short* Wt = (unsigned short*)(ph + NB * NU);

    k_w1_to_bf16_t<<<1024, 256, 0, stream>>>(W1, Wt);
    k_hidden_proj<<<NB, 256, 0, stream>>>(hidden, W2, b1, b2, ph);
    dim3 grid(NT / 64, NB);
    k_scores<<<grid, 256, 0, stream>>>(features, Wt, ph, V, bv, logits);
    k_softmax_ctx<<<NB, 256, 0, stream>>>(features, logits, (float*)d_out);
}
